// SimpleRNN_39865886441575
// MI455X (gfx1250) — matrix operand order repaired and checked by reading, not compiled, not run on hardware
//
#include <hip/hip_runtime.h>

// ---------------------------------------------------------------------------
// out[m,n] = tanh( x[m,:]·Wi[n,:] + bi[n] + bh[n] + h0·Wh[n,:] )
//   M = 131072, N = 512, K = 512; plus trailing tuple scalar 1.
// bf16 WMMA GEMM, f32 accumulate, bias folded into acc init, hw-tanh epilogue.
// A: register-prefetched (loads stay in flight across the barrier),
//    converted f32->bf16, stored to LDS.
// B: GLOBAL_LOAD_ASYNC_TO_LDS_B128 (no VGPR footprint, ASYNCcnt-tracked).
// Double-buffered LDS, one split-barrier per K-chunk (dscnt+asynccnt only —
// no LOADcnt drain, unlike __syncthreads()).
// ---------------------------------------------------------------------------

typedef __bf16 bf16_t;
typedef bf16_t v16bf __attribute__((ext_vector_type(16)));
typedef bf16_t v8bf  __attribute__((ext_vector_type(8)));
typedef float  v8f   __attribute__((ext_vector_type(8)));

#define M_TOTAL 131072
#define N_TOTAL 512
#define K_TOTAL 512
#define BM 128
#define BN 128
#define KC 64          // K-chunk staged in LDS (2 WMMA k-steps)
#define LDA 72         // LDS row pitch in halves (144B, keeps 16B alignment)
#define TILE_HALFS (BM * LDA)

#if __has_builtin(__builtin_amdgcn_tanhf)
#define FAST_TANH(x) __builtin_amdgcn_tanhf(x)
#elif __has_builtin(__builtin_amdgcn_tanh_f32)
#define FAST_TANH(x) __builtin_amdgcn_tanh_f32(x)
#else
#define FAST_TANH(x) tanhf(x)
#endif

// Producer->consumer tile barrier: waves must see each other's LDS writes
// (DScnt) and the async B DMA (ASYNCcnt), but private global loads (A
// prefetch) may stay in flight. __syncthreads() would drain LOADcnt too.
static __device__ __forceinline__ void tile_barrier() {
    asm volatile("s_wait_dscnt 0x0" ::: "memory");
    asm volatile("s_wait_asynccnt 0x0" ::: "memory");
#if __has_builtin(__builtin_amdgcn_s_barrier_signal) && \
    __has_builtin(__builtin_amdgcn_s_barrier_wait)
    __builtin_amdgcn_s_barrier_signal(-1);
    __builtin_amdgcn_s_barrier_wait(-1);
#else
    __syncthreads();
#endif
}

// Pack 4 floats -> 4 bf16 (round-half-up + v_perm_b32 byte-pack).
static __device__ __forceinline__ void store4bf(bf16_t* p, float4 f) {
    unsigned ax = __float_as_uint(f.x) + 0x8000u;
    unsigned ay = __float_as_uint(f.y) + 0x8000u;
    unsigned az = __float_as_uint(f.z) + 0x8000u;
    unsigned aw = __float_as_uint(f.w) + 0x8000u;
    uint2 u;
    u.x = __builtin_amdgcn_perm(ay, ax, 0x07060302u);  // {y_hi16, x_hi16}
    u.y = __builtin_amdgcn_perm(aw, az, 0x07060302u);  // {w_hi16, z_hi16}
    *reinterpret_cast<uint2*>(p) = u;                  // 8-byte store
}

// 16-half fragment from two 8-half (16B) LDS chunks.
static __device__ __forceinline__ v16bf ld_frag2(const bf16_t* p0, const bf16_t* p1) {
    v8bf lo = *reinterpret_cast<const v8bf*>(p0);
    v8bf hi = *reinterpret_cast<const v8bf*>(p1);
    v16bf r;
#pragma unroll
    for (int i = 0; i < 8; ++i) { r[i] = lo[i]; r[i + 8] = hi[i]; }
    return r;
}

// --- Kernel 1: fused bias c[n] = bi[n] + bh[n] + dot(h0, Wh[n,:]);
//     also writes the trailing tuple scalar out[M*N] = 1.
__global__ void rnn_bias_kernel(const float* __restrict__ Wh,
                                const float* __restrict__ bi,
                                const float* __restrict__ bh,
                                const float* __restrict__ h0,
                                float* __restrict__ cbias,
                                float* __restrict__ out_tail) {
    int h = threadIdx.x;                 // 512 threads
    float s = bi[h] + bh[h];
    for (int k = 0; k < K_TOTAL; k += 4) {
        float4 a = *reinterpret_cast<const float4*>(h0 + k);
        float4 w = *reinterpret_cast<const float4*>(Wh + (size_t)h * K_TOTAL + k);
        s += a.x * w.x + a.y * w.y + a.z * w.z + a.w * w.w;
    }
    cbias[h] = s;
    if (h == 0) out_tail[(size_t)M_TOTAL * N_TOTAL] = 1.0f;
}

// --- Kernel 2: Wi f32[512,512] -> bf16 (K-contiguous rows, WMMA-B order)
__global__ void wi_to_bf16_kernel(const float* __restrict__ Wi,
                                  bf16_t* __restrict__ wib) {
    int i = (blockIdx.x * blockDim.x + threadIdx.x) * 4;
    float4 f = *reinterpret_cast<const float4*>(Wi + i);
    store4bf(wib + i, f);
}

// --- Kernel 3: main GEMM + tanh epilogue
__global__ __launch_bounds__(256, 1)
void rnn_gemm_kernel(const float* __restrict__ x,
                     const bf16_t* __restrict__ wib,
                     const float* __restrict__ cbias,
                     float* __restrict__ out) {
    // [0]=A ping, [1]=A pong, [2]=B ping, [3]=B pong
    __shared__ __attribute__((aligned(128))) bf16_t smem[4 * TILE_HALFS];

    const int tid  = threadIdx.x;
    const int lane = tid & 31;
    const int wv   = tid >> 5;          // 8 waves
    const int mw   = wv >> 1;           // 0..3 : 32-row strip
    const int nw   = wv & 1;            // 0..1 : 64-col strip
    const int lnib = lane & 15;
    const int lhi  = lane >> 4;         // 0 or 1

    const int mBlock = blockIdx.y * BM;
    const int nBlock = blockIdx.x * BN;

    // Accumulators pre-loaded with the fused bias (C/D layout: lane&15 = N).
    v8f acc[2][4];
#pragma unroll
    for (int tj = 0; tj < 4; ++tj) {
        float b = cbias[nBlock + nw * 64 + tj * 16 + lnib];
#pragma unroll
        for (int ti = 0; ti < 2; ++ti)
#pragma unroll
            for (int v = 0; v < 8; ++v) acc[ti][tj][v] = b;
    }

    // Staging roles: thread t owns row (t>>1), half-chunk (t&1)*32 of KC=64.
    const int srow = tid >> 1;
    const int skh  = (tid & 1) * 32;
    const float* gxBase = x + (size_t)(mBlock + srow) * K_TOTAL + skh;

    // B async-copy parameters (per thread, 64 bytes per chunk).
    const unsigned long long wib_u64 = (unsigned long long)wib;
    const unsigned bGoffBase =
        (unsigned)(((nBlock + srow) * K_TOTAL + skh) * sizeof(bf16_t));
    // Generic-pointer low 32 bits == LDS byte address (ISA aperture rule).
    const unsigned bLds0 =
        (unsigned)(unsigned long long)(uintptr_t)(smem + 2 * TILE_HALFS + srow * LDA + skh);
    const unsigned bLds1 =
        (unsigned)(unsigned long long)(uintptr_t)(smem + 3 * TILE_HALFS + srow * LDA + skh);

    float4 aReg[8];
    // Prologue: fetch A chunk kc = 0 into registers.
#pragma unroll
    for (int q = 0; q < 8; ++q)
        aReg[q] = reinterpret_cast<const float4*>(gxBase)[q];

    for (int kc = 0; kc < K_TOTAL; kc += KC) {
        const int p = (kc >> 6) & 1;
        bf16_t* As = smem + p * TILE_HALFS;
        bf16_t* Bs = smem + (2 + p) * TILE_HALFS;

        // Kick off B tile DMA: global -> LDS, no VGPR data path.
        // offset:imm is added to BOTH global and LDS addresses, matching the
        // 16B-per-beat advance on both sides of the segment.
        {
            const unsigned vOff  = bGoffBase + (unsigned)(kc * sizeof(bf16_t));
            const unsigned ldsA  = p ? bLds1 : bLds0;
            asm volatile("global_load_async_to_lds_b128 %0, %1, %2 offset:0"
                         :: "v"(ldsA), "v"(vOff), "s"(wib_u64) : "memory");
            asm volatile("global_load_async_to_lds_b128 %0, %1, %2 offset:16"
                         :: "v"(ldsA), "v"(vOff), "s"(wib_u64) : "memory");
            asm volatile("global_load_async_to_lds_b128 %0, %1, %2 offset:32"
                         :: "v"(ldsA), "v"(vOff), "s"(wib_u64) : "memory");
            asm volatile("global_load_async_to_lds_b128 %0, %1, %2 offset:48"
                         :: "v"(ldsA), "v"(vOff), "s"(wib_u64) : "memory");
        }

        // Stage A chunk into LDS buffer p (f32 -> bf16 convert).
        {
            bf16_t* la = As + srow * LDA + skh;
#pragma unroll
            for (int q = 0; q < 8; ++q) store4bf(la + q * 4, aReg[q]);
        }

        // Issue next A chunk's global loads: they stay in flight across the
        // split barrier and overlap all 16 WMMAs below.
        if (kc + KC < K_TOTAL) {
            const float* gx = gxBase + kc + KC;
#pragma unroll
            for (int q = 0; q < 8; ++q)
                aReg[q] = reinterpret_cast<const float4*>(gx)[q];
        }

        tile_barrier();   // dscnt + asynccnt only; LOADcnt stays outstanding

#pragma unroll
        for (int kk = 0; kk < KC / 32; ++kk) {
            const int kh8 = lhi * 8;
            v16bf afr[2];
#pragma unroll
            for (int ti = 0; ti < 2; ++ti) {
                const bf16_t* pa =
                    As + (mw * 32 + ti * 16 + lnib) * LDA + kk * 32 + kh8;
                afr[ti] = ld_frag2(pa, pa + 16);
            }
#pragma unroll
            for (int tjp = 0; tjp < 2; ++tjp) {
                v16bf bfr[2];
#pragma unroll
                for (int j = 0; j < 2; ++j) {
                    const int tj = tjp * 2 + j;
                    const bf16_t* pb =
                        Bs + (nw * 64 + tj * 16 + lnib) * LDA + kk * 32 + lhi * 8;
                    bfr[j] = ld_frag2(pb, pb + 16);
                }
#pragma unroll
                for (int j = 0; j < 2; ++j)
#pragma unroll
                    for (int ti = 0; ti < 2; ++ti)
                        acc[ti][tjp * 2 + j] = __builtin_amdgcn_wmma_f32_16x16x32_bf16(
                            false, afr[ti], false, bfr[j], (short)0,
                            acc[ti][tjp * 2 + j], false, false);
            }
        }
    }

    // Epilogue: hardware tanh + streaming stores (output never re-read).
#pragma unroll
    for (int ti = 0; ti < 2; ++ti) {
        const int mrow = mBlock + mw * 32 + ti * 16 + lhi * 8;
#pragma unroll
        for (int tj = 0; tj < 4; ++tj) {
            const int ncol = nBlock + nw * 64 + tj * 16 + lnib;
#pragma unroll
            for (int v = 0; v < 8; ++v) {
                float r = FAST_TANH(acc[ti][tj][v]);
                __builtin_nontemporal_store(
                    r, out + (size_t)(mrow + v) * N_TOTAL + ncol);
            }
        }
    }
}

extern "C" void kernel_launch(void* const* d_in, const int* in_sizes, int n_in,
                              void* d_out, int out_size, void* d_ws, size_t ws_size,
                              hipStream_t stream) {
    (void)in_sizes; (void)n_in; (void)out_size; (void)ws_size;
    const float* x  = (const float*)d_in[0];   // [64,2048,512]
    const float* Wi = (const float*)d_in[1];   // [512,512]
    const float* bi = (const float*)d_in[2];   // [512]
    const float* Wh = (const float*)d_in[3];   // [512,512]
    const float* bh = (const float*)d_in[4];   // [512]
    const float* h0 = (const float*)d_in[5];   // [1,512]
    float* out = (float*)d_out;

    float*  cbias = (float*)d_ws;                       // 2 KB
    bf16_t* wib   = (bf16_t*)((char*)d_ws + 4096);      // 512 KB

    rnn_bias_kernel<<<1, 512, 0, stream>>>(Wh, bi, bh, h0, cbias, out);
    wi_to_bf16_kernel<<<(N_TOTAL * K_TOTAL / 4 + 255) / 256, 256, 0, stream>>>(Wi, wib);

    dim3 grid(N_TOTAL / BN, M_TOTAL / BM);   // (4, 1024)
    rnn_gemm_kernel<<<grid, 256, 0, stream>>>(x, wib, cbias, out);
}
